// FP4Linear_3702261809856
// MI455X (gfx1250) — 8-bit matrix operand order repaired and checked by reading, compiled, not run on hardware
//
#include <hip/hip_runtime.h>
#include <hip/hip_bf16.h>

typedef int   v16i __attribute__((ext_vector_type(16)));
typedef float v8f  __attribute__((ext_vector_type(8)));

__global__ __launch_bounds__(256) void absmax_kernel(const float* __restrict__ x,
                                                     size_t n, unsigned* __restrict__ slot) {
    __shared__ unsigned sdata[256];
    unsigned m = 0u;
    size_t i      = (size_t)blockIdx.x * blockDim.x + threadIdx.x;
    size_t stride = (size_t)gridDim.x * blockDim.x;
    for (; i < n; i += stride) m = max(m, __float_as_uint(fabsf(x[i])));
    sdata[threadIdx.x] = m;
    __syncthreads();
    for (int s = 128; s > 0; s >>= 1) {
        if ((int)threadIdx.x < s) sdata[threadIdx.x] = max(sdata[threadIdx.x], sdata[threadIdx.x + s]);
        __syncthreads();
    }
    if (threadIdx.x == 0) atomicMax(slot, sdata[0]);
}

__device__ __forceinline__ unsigned fp4_byte(float y) {
    float mag = fminf(fabsf(y), 6.0f);
    int idx = (mag > 0.25f) + (mag > 0.75f) + (mag > 1.25f) + (mag > 1.75f)
            + (mag > 2.5f)  + (mag > 3.5f)  + (mag > 5.0f);
    unsigned b = (unsigned)((0x4C4844403C383000ull >> (idx * 8)) & 0xFFull);
    if (y < 0.0f) b |= 0x80u;
    return b;
}

__device__ __forceinline__ float tensor_scale(unsigned bits) {
    return fmaxf(__uint_as_float(bits) * (1.0f / 6.0f), 1e-12f);
}

__global__ __launch_bounds__(256) void quant_pack_A(const float* __restrict__ x,
                                                    const unsigned* __restrict__ amax,
                                                    unsigned* __restrict__ apk,
                                                    int M, int K) {
    int idx = blockIdx.x * blockDim.x + threadIdx.x;
    int words = (M * K) >> 2;
    if (idx >= words) return;
    float inv = 1.0f / tensor_scale(*amax);

    int kwords = K >> 2;
    int gm = idx / kwords;
    int gk = (idx - gm * kwords) << 2;

    float4 v = ((const float4*)x)[idx];
    unsigned word =  fp4_byte(v.x * inv)
                  | (fp4_byte(v.y * inv) << 8)
                  | (fp4_byte(v.z * inv) << 16)
                  | (fp4_byte(v.w * inv) << 24);

    int m  = gm & 15,  tm = gm >> 4;
    int k  = gk & 127, tk = gk >> 7;
    int kk = k & 63;
    int vv = ((kk >> 5) << 2) + (((kk >> 3) & 1) << 1) + ((kk >> 2) & 1);
    int V  = ((k >> 6) << 3) + vv;
    int lane = (((kk >> 4) & 1) << 4) + m;
    size_t tile = (size_t)tm * (K >> 7) + tk;
    apk[tile * 512 + lane * 16 + V] = word;
}

__global__ __launch_bounds__(256) void quant_pack_B(const float* __restrict__ w,
                                                    const unsigned* __restrict__ amax,
                                                    unsigned* __restrict__ bpk,
                                                    int K, int N) {
    int idx = blockIdx.x * blockDim.x + threadIdx.x;
    int words = (K * N) >> 2;
    if (idx >= words) return;
    float inv = 1.0f / tensor_scale(*amax);

    int n  = idx % N;
    int kb = (idx / N) << 2;

    unsigned word = 0;
#pragma unroll
    for (int i = 0; i < 4; ++i)
        word |= fp4_byte(w[(size_t)(kb + i) * N + n] * inv) << (8 * i);

    int nn = n & 15,  tn = n >> 4;
    int k  = kb & 127, tk = kb >> 7;
    int g  = k >> 4;
    int kk = k & 15;
    int V  = ((g >> 1) << 2) + (kk >> 2);
    int lane = ((g & 1) << 4) + nn;
    size_t tile = (size_t)tn * (K >> 7) + tk;
    bpk[tile * 512 + lane * 16 + V] = word;
}

__device__ __forceinline__ void wmma_step(const unsigned char* __restrict__ base,
                                          int wm, int wn, int lane, v8f c[2][4]) {
    v16i a[2], b[4];
#pragma unroll
    for (int i = 0; i < 2; ++i)
        a[i] = *(const v16i*)(base + (((wm << 1) + i) << 11) + (lane << 6));
#pragma unroll
    for (int j = 0; j < 4; ++j)
        b[j] = *(const v16i*)(base + 16384 + (((wn << 2) + j) << 11) + (lane << 6));
#pragma unroll
    for (int i = 0; i < 2; ++i)
#pragma unroll
        for (int j = 0; j < 4; ++j)
            c[i][j] = __builtin_amdgcn_wmma_f32_16x16x128_fp8_fp8(
                          a[i], b[j], (short)0, c[i][j], false, false);
}

__global__ __launch_bounds__(256) void fp4_wmma_gemm(const unsigned char* __restrict__ apk,
                                                     const unsigned char* __restrict__ bpk,
                                                     const unsigned* __restrict__ scales,
                                                     const float* __restrict__ bias,
                                                     float* __restrict__ out,
                                                     int M, int N, int K) {
    __shared__ __align__(128) unsigned char smem[65536];

    const int KT   = K >> 7;
    const int lane = threadIdx.x & 31;
    const int wave = threadIdx.x >> 5;
    const int wm   = wave >> 1;
    const int wn   = wave & 1;

    const int tm0 = (blockIdx.x << 3);
    const int tn0 = (blockIdx.y << 3);

    const float s = tensor_scale(scales[0]) * tensor_scale(scales[1]);

    const v16i* gA = (const v16i*)(apk + (((size_t)(tm0 + wave) * KT) << 11) + (lane << 6));
    const v16i* gB = (const v16i*)(bpk + (((size_t)(tn0 + wave) * KT) << 11) + (lane << 6));
    const unsigned soff = (wave << 11) + (lane << 6);

    {
        v16i ra = gA[0], rb = gB[0];
        *(v16i*)(smem + soff)         = ra;
        *(v16i*)(smem + 16384 + soff) = rb;
    }

    v8f c[2][4] = {};
    for (int kt = 0; kt < KT; kt += 2) {
        __syncthreads();
        __builtin_prefetch(gA + (kt + 2) * 32, 0, 1);
        __builtin_prefetch(gB + (kt + 2) * 32, 0, 1);
        v16i ra = gA[(kt + 1) * 32];
        v16i rb = gB[(kt + 1) * 32];
        wmma_step(smem, wm, wn, lane, c);
        *(v16i*)(smem + 32768 + soff)         = ra;
        *(v16i*)(smem + 32768 + 16384 + soff) = rb;

        __syncthreads();
        const bool more = (kt + 2) < KT;
        v16i ra2, rb2;
        if (more) {
            ra2 = gA[(kt + 2) * 32];
            rb2 = gB[(kt + 2) * 32];
        }
        wmma_step(smem + 32768, wm, wn, lane, c);
        if (more) {
            *(v16i*)(smem + soff)         = ra2;
            *(v16i*)(smem + 16384 + soff) = rb2;
        }
    }

    const int half = lane >> 4;
    const int coln = lane & 15;
#pragma unroll
    for (int i = 0; i < 2; ++i) {
        int row0 = ((tm0 + (wm << 1) + i) << 4) + (half << 3);
#pragma unroll
        for (int j = 0; j < 4; ++j) {
            int col = ((tn0 + (wn << 2) + j) << 4) + coln;
            float bb = bias[col];
#pragma unroll
            for (int r = 0; r < 8; ++r)
                out[(size_t)(row0 + r) * N + col] = c[i][j][r] * s + bb;
        }
    }
}

extern "C" void kernel_launch(void* const* d_in, const int* in_sizes, int n_in,
                              void* d_out, int out_size, void* d_ws, size_t ws_size,
                              hipStream_t stream) {
    const float* x    = (const float*)d_in[0];
    const float* w    = (const float*)d_in[1];
    const float* bias = (const float*)d_in[2];
    float*       out  = (float*)d_out;

    const int N = in_sizes[2];
    const int K = in_sizes[1] / N;
    const int M = in_sizes[0] / K;

    unsigned char* ws     = (unsigned char*)d_ws;
    unsigned*      scales = (unsigned*)ws;
    unsigned char* apk    = ws + 1024;
    unsigned char* bpk    = apk + (size_t)M * K;

    hipMemsetAsync(scales, 0, 256, stream);
    absmax_kernel<<<1024, 256, 0, stream>>>(x, (size_t)M * K, scales + 0);
    absmax_kernel<<<512,  256, 0, stream>>>(w, (size_t)K * N, scales + 1);

    int wordsA = (M * K) >> 2;
    int wordsB = (K * N) >> 2;
    quant_pack_A<<<(wordsA + 255) / 256, 256, 0, stream>>>(x, scales + 0, (unsigned*)apk, M, K);
    quant_pack_B<<<(wordsB + 255) / 256, 256, 0, stream>>>(w, scales + 1, (unsigned*)bpk, K, N);

    dim3 grid(M / 128, N / 128);
    fp4_wmma_gemm<<<grid, 256, 0, stream>>>(apk, bpk, scales, bias, out, M, N, K);
}
